// TopkLinear_47304769798214
// MI455X (gfx1250) — matrix operand order repaired and checked by reading, not compiled, not run on hardware
//
#include <hip/hip_runtime.h>
#include <hip/hip_bf16.h>
#include <stdint.h>

// ---------------- CDNA5 types ----------------
typedef __attribute__((ext_vector_type(8)))  float          v8f;
typedef __attribute__((ext_vector_type(16))) __bf16         v16bf;
typedef __attribute__((ext_vector_type(4)))  unsigned int   u32x4;
typedef __attribute__((ext_vector_type(8)))  int            i32x8;
typedef __attribute__((ext_vector_type(4)))  int            i32x4;
typedef __attribute__((ext_vector_type(4)))  float          f32x4;
typedef __attribute__((ext_vector_type(4)))  unsigned short u16x4;

#define IN_F   2048
#define OUT_F  8192
#define NROWS  8192
#define NBI    64          // in-feature blocks (K blocks of 32)
#define NBJ    64          // out-feature blocks (N blocks of 128)
#define KBLK   32
#define NBLK   128
#define MT     128         // M tile per workgroup
#define TOPK_K 410

#if __has_builtin(__builtin_amdgcn_tensor_load_to_lds)
  #define HAVE_TDM 1
  #if __has_include(<hip/amd_detail/amd_gfx1250_TDM.h>)
    #define TDM_SIX_ARG 1
  #endif
#endif

union FragBF { v16bf v; u32x4 q[2]; };

__device__ __forceinline__ unsigned short f2bf(float f) {
  union { float f; unsigned u; } v; v.f = f;
  unsigned u = v.u;
  u += 0x7FFFu + ((u >> 16) & 1u);   // round-to-nearest-even
  return (unsigned short)(u >> 16);
}

// ---------------- K1: block avg-pool of |W| ----------------
// pooled[bi*64+bj] = mean_{o in [bj*128,+128), k in [bi*32,+32)} |W[o][k]|
__global__ __launch_bounds__(256) void pool_kernel(const float* __restrict__ W,
                                                   float* __restrict__ pooled) {
  __shared__ float red[256];
  const int blk = blockIdx.x;          // 0..4095
  const int bi = blk >> 6, bj = blk & 63;
  const float* base = W + (size_t)bj * NBLK * IN_F + bi * KBLK;
  float s = 0.f;
  #pragma unroll
  for (int i = 0; i < 16; ++i) {
    const int e = threadIdx.x + 256 * i;   // 0..4095
    const int o = e >> 5, kk = e & 31;
    s += fabsf(base[(size_t)o * IN_F + kk]);
  }
  red[threadIdx.x] = s;
  __syncthreads();
  for (int off = 128; off > 0; off >>= 1) {
    if ((int)threadIdx.x < off) red[threadIdx.x] += red[threadIdx.x + off];
    __syncthreads();
  }
  if (threadIdx.x == 0) pooled[blk] = red[0] * (1.0f / 4096.0f);
}

// ---------------- K2: exact top-k via rank (jax top_k tie-break: lower index wins) ----
__global__ __launch_bounds__(256) void topk_mask_kernel(const float* __restrict__ pooled,
                                                        int* __restrict__ flag) {
  __shared__ float sp[NBI * NBJ];
  for (int i = threadIdx.x; i < NBI * NBJ; i += 256) sp[i] = pooled[i];
  __syncthreads();
  for (int i = threadIdx.x; i < NBI * NBJ; i += 256) {
    const float v = sp[i];
    int rank = 0;
    for (int j = 0; j < NBI * NBJ; ++j) {
      const float u = sp[j];
      rank += (u > v) || (u == v && j < i);
    }
    flag[i] = (rank < TOPK_K) ? 1 : 0;
  }
}

// ---------------- K3: per-bj list of active bi blocks ----------------
__global__ void compact_kernel(const int* __restrict__ flag,
                               int* __restrict__ counts, int* __restrict__ lists) {
  const int bj = threadIdx.x;
  if (bj >= NBJ) return;
  int c = 0;
  for (int bi = 0; bi < NBI; ++bi)
    if (flag[bi * NBJ + bj]) lists[bj * NBI + (c++)] = bi;
  counts[bj] = c;
}

// ---------------- K4: convert x to bf16 ----------------
__global__ __launch_bounds__(256) void cvt_x_kernel(const float* __restrict__ x,
                                                    unsigned short* __restrict__ xb) {
  const size_t i = ((size_t)blockIdx.x * 256 + threadIdx.x) * 4;
  f32x4 v = *(const f32x4*)(x + i);
  u16x4 o;
  o.x = f2bf(v.x); o.y = f2bf(v.y); o.z = f2bf(v.z); o.w = f2bf(v.w);
  *(u16x4*)(xb + i) = o;
}

// ---------------- K5: convert only active W blocks to bf16 ----------------
__global__ __launch_bounds__(256) void cvt_w_kernel(const float* __restrict__ W,
                                                    const int* __restrict__ flag,
                                                    unsigned short* __restrict__ wb) {
  const int blk = blockIdx.x;
  if (!flag[blk]) return;
  const int bi = blk >> 6, bj = blk & 63;
  const size_t base = (size_t)bj * NBLK * IN_F + bi * KBLK;
  #pragma unroll
  for (int i = 0; i < 16; ++i) {
    const int e = threadIdx.x + 256 * i;
    const int o = e >> 5, kk = e & 31;
    const size_t idx = base + (size_t)o * IN_F + kk;
    wb[idx] = f2bf(W[idx]);
  }
}

// ---------------- TDM helper: 2D bf16 tile -> LDS ----------------
#ifdef HAVE_TDM
__device__ __forceinline__ void tdm_load_2d_bf16(unsigned lds_off, const void* gptr,
                                                 unsigned tile0, unsigned tile1,
                                                 unsigned long long stride0_elems) {
  const unsigned long long ga = (unsigned long long)(uintptr_t)gptr;
  u32x4 g0;
  g0.x = 1u;                                            // count=1, user descriptor
  g0.y = lds_off;                                       // LDS byte address
  g0.z = (unsigned)ga;                                  // global_addr[31:0]
  g0.w = (unsigned)((ga >> 32) & 0x1FFFFFFu) | (2u << 30); // addr[56:32] | type=2
  const unsigned tdim0 = tile0, tdim1 = tile1;          // tensor dims == tile (no OOB)
  i32x8 g1;
  g1[0] = (int)(1u << 16);                              // data_size=1 (2 bytes), mask=0
  g1[1] = (int)((tdim0 & 0xFFFFu) << 16);               // [63:48] = tensor_dim0 lo16
  g1[2] = (int)((tdim0 >> 16) | ((tdim1 & 0xFFFFu) << 16));
  g1[3] = (int)((tdim1 >> 16) | (tile0 << 16));         // [127:112] = tile_dim0
  g1[4] = (int)(tile1 & 0xFFFFu);                       // tile_dim1, tile_dim2=0
  g1[5] = (int)(unsigned)(stride0_elems & 0xFFFFFFFFull);
  g1[6] = (int)(unsigned)((stride0_elems >> 32) & 0xFFFFull);
  g1[7] = 0;
  i32x4 z4 = {0, 0, 0, 0};
  #ifdef TDM_SIX_ARG
  i32x8 z8 = {0, 0, 0, 0, 0, 0, 0, 0};
  __builtin_amdgcn_tensor_load_to_lds(g0, g1, z4, z4, z8, 0);
  #else
  __builtin_amdgcn_tensor_load_to_lds(g0, g1, z4, z4, 0);
  #endif
}
#endif

// ---------------- K6: block-sparse GEMM, 128x128 tile / WG, 8 waves ----------------
// y[m, bj*128+n] = sum over active bi: x[m, bi*32:+32] . W[bj*128+n, bi*32:+32] + b
__global__ __launch_bounds__(256) void gemm_kernel(const unsigned short* __restrict__ xbf,
                                                   const unsigned short* __restrict__ wbf,
                                                   const float* __restrict__ bias,
                                                   const int* __restrict__ counts,
                                                   const int* __restrict__ lists,
                                                   float* __restrict__ out) {
  __shared__ __attribute__((aligned(16))) unsigned short xs[MT][KBLK];   // 8 KB
  __shared__ __attribute__((aligned(16))) unsigned short wsh[NBLK][KBLK]; // 8 KB

  const int mtile = blockIdx.x, bj = blockIdx.y;
  const int m0 = mtile * MT;
  const int tid = threadIdx.x;
  const int lane = tid & 31, wave = tid >> 5;
  const int wr = wave & 3;      // 32-row group within tile
  const int wc = wave >> 1 >> 1;// 64-col group within tile (wave/4)
  const int l16 = lane & 15;

  v8f acc[2][4] = {};
  const int cnt = counts[bj];

  for (int it = 0; it < cnt; ++it) {
    const int bi = lists[bj * NBI + it];
    __syncthreads();   // LDS reads of previous iter done
#ifdef HAVE_TDM
    if (wave == 0) {
      tdm_load_2d_bf16((unsigned)(size_t)&xs[0][0],
                       xbf + (size_t)m0 * IN_F + bi * KBLK, KBLK, MT, IN_F);
      tdm_load_2d_bf16((unsigned)(size_t)&wsh[0][0],
                       wbf + (size_t)bj * NBLK * IN_F + bi * KBLK, KBLK, NBLK, IN_F);
      __builtin_amdgcn_s_wait_tensorcnt(0);
    }
#else
    for (int c = tid; c < 512; c += 256) {   // 512 x 16B per tile
      const int row = c >> 2;
      const int off = (c & 3) << 3;          // half-element offset
      *(u32x4*)&xs[row][off] =
          *(const u32x4*)(xbf + (size_t)(m0 + row) * IN_F + bi * KBLK + off);
      *(u32x4*)&wsh[row][off] =
          *(const u32x4*)(wbf + (size_t)(bj * NBLK + row) * IN_F + bi * KBLK + off);
    }
#endif
    __syncthreads();

    // A fragments (16x32 bf16): lanes 0-15 row M=l16 hold K{0..7,16..23},
    // lanes 16-31 hold K{8..15,24..31} (ISA 7.12.2)
    const int ka = (lane < 16) ? 0 : 8;
    FragBF a[2];
    #pragma unroll
    for (int mi = 0; mi < 2; ++mi) {
      const int m = wr * 32 + mi * 16 + l16;
      a[mi].q[0] = *(const u32x4*)&xs[m][ka];
      a[mi].q[1] = *(const u32x4*)&xs[m][ka + 16];
    }
    // B fragments (32x16 bf16): lanes 0-15 col N=l16 hold K{0..7,16..23}, lanes 16-31 K{8..15,24..31}
    const int kb = (lane < 16) ? 0 : 8;
    #pragma unroll
    for (int ni = 0; ni < 4; ++ni) {
      const int n = wc * 64 + ni * 16 + l16;
      FragBF bfr;
      bfr.q[0] = *(const u32x4*)&wsh[n][kb];
      bfr.q[1] = *(const u32x4*)&wsh[n][kb + 16];
      #pragma unroll
      for (int mi = 0; mi < 2; ++mi) {
        acc[mi][ni] = __builtin_amdgcn_wmma_f32_16x16x32_bf16(
            false, a[mi].v, false, bfr.v, (short)0, acc[mi][ni], false, false);
      }
    }
  }

  // Epilogue: C layout — lanes 0-15: N=l16, VGPR r -> M=r; lanes 16-31: M=r+8
  #pragma unroll
  for (int mi = 0; mi < 2; ++mi) {
    #pragma unroll
    for (int ni = 0; ni < 4; ++ni) {
      const int ncol = bj * NBLK + wc * 64 + ni * 16 + l16;
      const float bv = bias[ncol];
      const int mbase = m0 + wr * 32 + mi * 16 + ((lane < 16) ? 0 : 8);
      #pragma unroll
      for (int r = 0; r < 8; ++r)
        out[(size_t)(mbase + r) * OUT_F + ncol] = acc[mi][ni][r] + bv;
    }
  }
}

// ---------------- launch ----------------
extern "C" void kernel_launch(void* const* d_in, const int* in_sizes, int n_in,
                              void* d_out, int out_size, void* d_ws, size_t ws_size,
                              hipStream_t stream) {
  const float* x = (const float*)d_in[0];   // (8192, 2048)
  const float* W = (const float*)d_in[1];   // (8192, 2048)
  const float* b = (const float*)d_in[2];   // (8192,)
  float* out = (float*)d_out;               // (8192, 8192)

  char* ws = (char*)d_ws;
  float*          pooled = (float*)(ws);                         // 16 KB
  int*            flag   = (int*)(ws + 16384);                   // 16 KB
  int*            counts = (int*)(ws + 32768);                   // 256 B
  int*            lists  = (int*)(ws + 33024);                   // 16 KB
  unsigned short* xbf    = (unsigned short*)(ws + 65536);        // 32 MB
  unsigned short* wbf    = (unsigned short*)(ws + 65536 + (size_t)NROWS * IN_F * 2);

  pool_kernel<<<NBI * NBJ, 256, 0, stream>>>(W, pooled);
  topk_mask_kernel<<<1, 256, 0, stream>>>(pooled, flag);
  compact_kernel<<<1, 64, 0, stream>>>(flag, counts, lists);
  cvt_x_kernel<<<(int)(((size_t)NROWS * IN_F) / (256 * 4)), 256, 0, stream>>>(x, xbf);
  cvt_w_kernel<<<NBI * NBJ, 256, 0, stream>>>(W, flag, wbf);
  gemm_kernel<<<dim3(NROWS / MT, NBJ), 256, 0, stream>>>(xbf, wbf, b, counts, lists, out);
}
